// WaveletPCTransformer_5188320493665
// MI455X (gfx1250) — compile-verified, matrix operand order checked by reading, not run on hardware
//
#include <hip/hip_runtime.h>
#include <hip/hip_bf16.h>

// ---------------------------------------------------------------------------
// WaveletLinear forward for MI455X (gfx1250, wave32, WMMA).
//   B=16, S=2048, D=1024  -> R = 32768 rows.
//   Stage 1: split weights to bf16 hi/lo
//   Stage 2: Haar DWT (closed form) -> coeff planes (bf16 hi/lo), row layout:
//            [ d0 (512) | d1 (256) | d2 (128) | a (128) ]  per 1024-wide row
//   Stage 3: per-level GEMM y = c @ W^T + b via v_wmma_f32_16x16x32_bf16,
//            bf16x3 emulation (hi*hi + hi*lo + lo*hi) for ~fp32 accuracy
//   Stage 4: Haar IDWT -> out (fp32)
// ---------------------------------------------------------------------------

typedef __attribute__((ext_vector_type(16))) __bf16 bf16x16;
typedef __attribute__((ext_vector_type(8)))  float  f32x8;

#define ROWS 32768
#define DIM  1024
#define INV_SQRT2 0.70710678118654752440f

// coeff-plane column offsets per level
#define OFF_D0 0
#define OFF_D1 512
#define OFF_D2 768
#define OFF_A  896

// packed weight-plane element offsets
#define WOFF_D0 0
#define WOFF_D1 262144   // 512*512
#define WOFF_D2 327680   // +256*256
#define WOFF_A  344064   // +128*128
#define WTOT    360448   // +128*128

// workspace byte offsets
#define WS_CHI  0ull
#define WS_CLO  67108864ull                 // 32768*1024*2
#define WS_Y    134217728ull                // + same
#define WS_WHI  268435456ull                // + 32768*1024*4
#define WS_WLO  (268435456ull + 720896ull)  // + WTOT*2

__device__ __forceinline__ unsigned short f2bf_rn(float f) {
    unsigned u = __float_as_uint(f);
    u += 0x7FFFu + ((u >> 16) & 1u);
    return (unsigned short)(u >> 16);
}

__device__ __forceinline__ void split_bf(float f, unsigned short& h, unsigned short& l) {
    h = f2bf_rn(f);
    float fh = __uint_as_float((unsigned)h << 16);
    l = f2bf_rn(f - fh);
}

union FragU {
    uint4   q[2];
    bf16x16 v;
};

// Fragment = two contiguous 16B chunks, 32B apart (elements p..p+7, p+16..p+23)
__device__ __forceinline__ bf16x16 load_frag(const unsigned short* p) {
    FragU f;
    f.q[0] = *reinterpret_cast<const uint4*>(p);
    f.q[1] = *reinterpret_cast<const uint4*>(p + 16);
    return f.v;
}

// ---------------------------------------------------------------------------
// Stage 1: split the four weight matrices into bf16 hi/lo planes
// ---------------------------------------------------------------------------
__global__ void __launch_bounds__(256)
wsplit_kernel(const float* __restrict__ W0, const float* __restrict__ W1,
              const float* __restrict__ W2, const float* __restrict__ Wa,
              unsigned short* __restrict__ w_hi, unsigned short* __restrict__ w_lo) {
    int idx = blockIdx.x * 256 + threadIdx.x;
    if (idx >= WTOT) return;
    float f;
    if (idx < WOFF_D1)      f = W0[idx];
    else if (idx < WOFF_D2) f = W1[idx - WOFF_D1];
    else if (idx < WOFF_A)  f = W2[idx - WOFF_D2];
    else                    f = Wa[idx - WOFF_A];
    unsigned short h, l;
    split_bf(f, h, l);
    w_hi[idx] = h;
    w_lo[idx] = l;
}

// ---------------------------------------------------------------------------
// Stage 2: Haar DWT, 3 levels, closed form per 8 consecutive inputs
// block = 256 threads = 2 rows (128 threads/row, 8 floats/thread)
// ---------------------------------------------------------------------------
__global__ void __launch_bounds__(256)
dwt_kernel(const float* __restrict__ x,
           unsigned short* __restrict__ c_hi, unsigned short* __restrict__ c_lo) {
    const float R = INV_SQRT2;
    int tid = threadIdx.x;
    int t   = tid & 127;                       // chunk within row
    int row = blockIdx.x * 2 + (tid >> 7);

    const float4* xp = reinterpret_cast<const float4*>(x + (size_t)row * DIM + t * 8);
    float4 xa = xp[0];
    float4 xb = xp[1];

    float d0_0 = (xa.x - xa.y) * R, d0_1 = (xa.z - xa.w) * R;
    float d0_2 = (xb.x - xb.y) * R, d0_3 = (xb.z - xb.w) * R;
    float a0_0 = (xa.x + xa.y) * R, a0_1 = (xa.z + xa.w) * R;
    float a0_2 = (xb.x + xb.y) * R, a0_3 = (xb.z + xb.w) * R;

    float d1_0 = (a0_0 - a0_1) * R, d1_1 = (a0_2 - a0_3) * R;
    float a1_0 = (a0_0 + a0_1) * R, a1_1 = (a0_2 + a0_3) * R;

    float d2 = (a1_0 - a1_1) * R;
    float aa = (a1_0 + a1_1) * R;

    unsigned short* ch = c_hi + (size_t)row * DIM;
    unsigned short* cl = c_lo + (size_t)row * DIM;

    unsigned short h0, l0, h1, l1, h2, l2, h3, l3;
    // d0: 4 values at columns 4t..4t+3
    split_bf(d0_0, h0, l0); split_bf(d0_1, h1, l1);
    split_bf(d0_2, h2, l2); split_bf(d0_3, h3, l3);
    uint2 ph = make_uint2((unsigned)h0 | ((unsigned)h1 << 16), (unsigned)h2 | ((unsigned)h3 << 16));
    uint2 pl = make_uint2((unsigned)l0 | ((unsigned)l1 << 16), (unsigned)l2 | ((unsigned)l3 << 16));
    *reinterpret_cast<uint2*>(ch + OFF_D0 + 4 * t) = ph;
    *reinterpret_cast<uint2*>(cl + OFF_D0 + 4 * t) = pl;
    // d1: 2 values at 512 + 2t
    split_bf(d1_0, h0, l0); split_bf(d1_1, h1, l1);
    *reinterpret_cast<unsigned*>(ch + OFF_D1 + 2 * t) = (unsigned)h0 | ((unsigned)h1 << 16);
    *reinterpret_cast<unsigned*>(cl + OFF_D1 + 2 * t) = (unsigned)l0 | ((unsigned)l1 << 16);
    // d2, a: 1 value each
    split_bf(d2, h0, l0);
    ch[OFF_D2 + t] = h0; cl[OFF_D2 + t] = l0;
    split_bf(aa, h1, l1);
    ch[OFF_A + t] = h1; cl[OFF_A + t] = l1;
}

// ---------------------------------------------------------------------------
// Stage 3: per-level GEMM  y[r, v] = sum_h c[r,h] * W[v,h] + b[v]
// grid.x = 256 row-tiles (128 rows each); grid.y = 16 column strips:
//   0..7  -> level 0 (K=512), 8..11 -> level 1 (K=256),
//   12..13-> level 2 (K=128), 14..15 -> approx (K=128)
// 8 waves/block; each wave: 16 rows x 64 cols (4 wmma tiles), bf16x3.
// ---------------------------------------------------------------------------
__global__ void __launch_bounds__(256)
gemm_kernel(const unsigned short* __restrict__ c_hi,
            const unsigned short* __restrict__ c_lo,
            const unsigned short* __restrict__ w_hi,
            const unsigned short* __restrict__ w_lo,
            float* __restrict__ y,
            const float* __restrict__ b0, const float* __restrict__ b1,
            const float* __restrict__ b2, const float* __restrict__ ba) {
    int strip = blockIdx.y;
    int K, coff, woff, colL;
    const float* bias;
    if (strip < 8)       { K = 512; coff = OFF_D0; woff = WOFF_D0; colL = strip * 64;        bias = b0; }
    else if (strip < 12) { K = 256; coff = OFF_D1; woff = WOFF_D1; colL = (strip - 8) * 64;  bias = b1; }
    else if (strip < 14) { K = 128; coff = OFF_D2; woff = WOFF_D2; colL = (strip - 12) * 64; bias = b2; }
    else                 { K = 128; coff = OFF_A;  woff = WOFF_A;  colL = (strip - 14) * 64; bias = ba; }

    int wave = threadIdx.x >> 5;
    int lane = threadIdx.x & 31;
    int rbase = blockIdx.x * 128 + wave * 16;

    int m     = lane & 15;          // A row within tile / B column within tile
    int koff8 = (lane >> 4) << 3;   // 0 or 8: k sub-chunk select

    const unsigned short* aH = c_hi + (size_t)(rbase + m) * DIM + coff + koff8;
    const unsigned short* aL = c_lo + (size_t)(rbase + m) * DIM + coff + koff8;
    const unsigned short* wH = w_hi + woff;
    const unsigned short* wL = w_lo + woff;

    f32x8 acc[4] = {};

    for (int kb = 0; kb < K; kb += 32) {
        bf16x16 Ah = load_frag(aH + kb);
        bf16x16 Al = load_frag(aL + kb);
#pragma unroll
        for (int tile = 0; tile < 4; ++tile) {
            size_t boff = (size_t)(colL + 16 * tile + m) * K + kb + koff8;
            bf16x16 Bh = load_frag(wH + boff);
            bf16x16 Bl = load_frag(wL + boff);
            acc[tile] = __builtin_amdgcn_wmma_f32_16x16x32_bf16(
                false, Ah, false, Bh, (short)0, acc[tile], false, false);
            acc[tile] = __builtin_amdgcn_wmma_f32_16x16x32_bf16(
                false, Ah, false, Bl, (short)0, acc[tile], false, false);
            acc[tile] = __builtin_amdgcn_wmma_f32_16x16x32_bf16(
                false, Al, false, Bh, (short)0, acc[tile], false, false);
        }
    }

    // Epilogue: C/D layout -> lanes 0-15: N=lane, M=r; lanes 16-31: N=lane-16, M=r+8
    int n  = lane & 15;
    int ms = (lane >> 4) * 8;
#pragma unroll
    for (int tile = 0; tile < 4; ++tile) {
        int v = colL + 16 * tile + n;      // column within level
        float bv = bias[v];
#pragma unroll
        for (int r = 0; r < 8; ++r) {
            y[(size_t)(rbase + r + ms) * DIM + coff + v] = acc[tile][r] + bv;
        }
    }
}

// ---------------------------------------------------------------------------
// Stage 4: Haar IDWT (exact synthesis), 8 outputs per thread
// ---------------------------------------------------------------------------
__global__ void __launch_bounds__(256)
idwt_kernel(const float* __restrict__ y, float* __restrict__ out) {
    const float R = INV_SQRT2;
    int tid = threadIdx.x;
    int t   = tid & 127;
    int row = blockIdx.x * 2 + (tid >> 7);

    const float* yr = y + (size_t)row * DIM;

    float4 y0 = *reinterpret_cast<const float4*>(yr + OFF_D0 + 4 * t);
    float2 y1 = *reinterpret_cast<const float2*>(yr + OFF_D1 + 2 * t);
    float  y2 = yr[OFF_D2 + t];
    float  ya = yr[OFF_A + t];

    // level-2 synthesis: a1_rec[2t], a1_rec[2t+1]
    float u0 = (ya + y2) * R;
    float u1 = (ya - y2) * R;
    // level-1 synthesis: a0_rec[4t .. 4t+3]
    float v0 = (u0 + y1.x) * R;
    float v1 = (u0 - y1.x) * R;
    float v2 = (u1 + y1.y) * R;
    float v3 = (u1 - y1.y) * R;
    // level-0 synthesis: out[8t .. 8t+7]
    float4 oa, ob;
    oa.x = (v0 + y0.x) * R; oa.y = (v0 - y0.x) * R;
    oa.z = (v1 + y0.y) * R; oa.w = (v1 - y0.y) * R;
    ob.x = (v2 + y0.z) * R; ob.y = (v2 - y0.z) * R;
    ob.z = (v3 + y0.w) * R; ob.w = (v3 - y0.w) * R;

    float4* op = reinterpret_cast<float4*>(out + (size_t)row * DIM + 8 * t);
    op[0] = oa;
    op[1] = ob;
}

// ---------------------------------------------------------------------------
extern "C" void kernel_launch(void* const* d_in, const int* in_sizes, int n_in,
                              void* d_out, int out_size, void* d_ws, size_t ws_size,
                              hipStream_t stream) {
    const float* x  = (const float*)d_in[0];
    const float* W0 = (const float*)d_in[1];
    const float* b0 = (const float*)d_in[2];
    const float* W1 = (const float*)d_in[3];
    const float* b1 = (const float*)d_in[4];
    const float* W2 = (const float*)d_in[5];
    const float* b2 = (const float*)d_in[6];
    const float* Wa = (const float*)d_in[7];
    const float* ba = (const float*)d_in[8];
    float* out = (float*)d_out;

    char* ws = (char*)d_ws;
    unsigned short* c_hi = (unsigned short*)(ws + WS_CHI);
    unsigned short* c_lo = (unsigned short*)(ws + WS_CLO);
    float*          yb   = (float*)(ws + WS_Y);
    unsigned short* w_hi = (unsigned short*)(ws + WS_WHI);
    unsigned short* w_lo = (unsigned short*)(ws + WS_WLO);

    // Stage 1: weight split (WTOT = 1408 * 256 exactly)
    wsplit_kernel<<<1408, 256, 0, stream>>>(W0, W1, W2, Wa, w_hi, w_lo);

    // Stage 2: DWT (2 rows per block)
    dwt_kernel<<<ROWS / 2, 256, 0, stream>>>(x, c_hi, c_lo);

    // Stage 3: all four level GEMMs in one grid
    gemm_kernel<<<dim3(ROWS / 128, 16), 256, 0, stream>>>(
        c_hi, c_lo, w_hi, w_lo, yb, b0, b1, b2, ba);

    // Stage 4: IDWT
    idwt_kernel<<<ROWS / 2, 256, 0, stream>>>(yb, out);
}
